// InstructionEncoder_29858612642554
// MI455X (gfx1250) — matrix operand order repaired and checked by reading, not compiled, not run on hardware
//
#include <hip/hip_runtime.h>
#include <hip/hip_fp16.h>

typedef _Float16 v16h __attribute__((ext_vector_type(16)));
typedef _Float16 v8h  __attribute__((ext_vector_type(8)));
typedef float    v8f  __attribute__((ext_vector_type(8)));

#define B_      64
#define H_      512
#define T_      512
#define I_      300
#define IP_     320            // I padded to multiple of 32
#define G4_     2048           // 4*H
#define KT_     832            // H + IP (fused K)
#define NWG_    32             // persistent workgroups
#define THR_    128            // 4 waves per WG

// ---- workspace layout (bytes) ----
#define WF_OFF    0ull                            // f16 [2048][832] fused [W_hh | W_ih | 0]
#define WF_BYTES  ((unsigned long long)G4_*KT_*2)
#define XT_OFF    (WF_OFF + WF_BYTES)             // f16 [T][B][IP] transposed/padded x
#define XT_BYTES  ((unsigned long long)T_*B_*IP_*2)
#define BIAS_OFF  (XT_OFF + XT_BYTES)             // f32 [2048] b_ih + b_hh
#define BIAS_BYTES ((unsigned long long)G4_*4)
#define H16_OFF   (BIAS_OFF + BIAS_BYTES)         // f16 [2][B][H] double-buffered h
#define H16_BYTES ((unsigned long long)2*B_*H_*2)
#define CNT_OFF   (H16_OFF + H16_BYTES)           // u32 [T] per-step arrival counters
#define CNT_BYTES ((unsigned long long)T_*4)

// ---------------- prep: fused f16 weights + bias ----------------
__global__ __launch_bounds__(256) void lstm_prep_weights(
    const float* __restrict__ Whh, const float* __restrict__ Wih,
    const float* __restrict__ bih, const float* __restrict__ bhh,
    _Float16* __restrict__ Wf, float* __restrict__ bias) {
  int idx = blockIdx.x * blockDim.x + threadIdx.x;   // over 2048*832
  if (idx >= G4_ * KT_) return;
  int n = idx / KT_, k = idx - n * KT_;
  float v = 0.f;
  if (k < H_)            v = Whh[n * H_ + k];
  else if (k < H_ + I_)  v = Wih[n * I_ + (k - H_)];
  Wf[idx] = (_Float16)v;
  if (k == 0) bias[n] = bih[n] + bhh[n];
}

// ---------------- prep: transpose x[B][I][T] -> xt[T][B][IP] (f16, padded) ----------------
__global__ __launch_bounds__(256) void lstm_prep_x(
    const float* __restrict__ x, _Float16* __restrict__ xt) {
  long long idx = (long long)blockIdx.x * blockDim.x + threadIdx.x; // over T*B*IP
  if (idx >= (long long)T_ * B_ * IP_) return;
  int i = (int)(idx % IP_);
  long long r = idx / IP_;
  int b = (int)(r % B_);
  int t = (int)(r / B_);
  float v = (i < I_) ? x[((long long)b * I_ + i) * T_ + t] : 0.f;
  xt[idx] = (_Float16)v;
}

// ---------------- device helpers ----------------
__device__ __forceinline__ float sigmoidf_(float x) { return 1.f / (1.f + __expf(-x)); }
__device__ __forceinline__ float tanhf_(float x)    { return 1.f - 2.f / (__expf(2.f * x) + 1.f); }

// A operand (16x32 f16): lane<16 -> row M=m0+lane, K = {k0..k0+7, k0+16..k0+23}
//                        lane>=16 -> same row set, K shifted by 8 (ISA 16-bit A layout)
__device__ __forceinline__ v16h loadA16x32(const _Float16* __restrict__ base,
                                           int stride, int m, int k0, int hiHalf) {
  const _Float16* p = base + (size_t)m * stride + k0 + (hiHalf << 3);
  v8h lo = *(const v8h*)p;
  v8h hi = *(const v8h*)(p + 16);
  return __builtin_shufflevector(lo, hi, 0,1,2,3,4,5,6,7,8,9,10,11,12,13,14,15);
}

// B operand (32x16 f16): lane<16 -> col N=lane, K = {k0..k0+7, k0+16..k0+23} ; lane>=16 -> K shifted by 8
__device__ __forceinline__ v16h loadB32x16_lds(const _Float16* __restrict__ ldsRow,
                                               int k0, int hiHalf) {
  const _Float16* p = ldsRow + k0 + (hiHalf << 3);
  v8h lo = *(const v8h*)p;
  v8h hi = *(const v8h*)(p + 16);
  return __builtin_shufflevector(lo, hi, 0,1,2,3,4,5,6,7,8,9,10,11,12,13,14,15);
}

#define WMMA_F16(a, b, c) \
  __builtin_amdgcn_wmma_f32_16x16x32_f16(false, (a), false, (b), (short)0, (c), false, false)

__device__ __forceinline__ void wait_asynccnt0() {
#if __has_builtin(__builtin_amdgcn_s_wait_asynccnt)
  __builtin_amdgcn_s_wait_asynccnt(0);
#else
  asm volatile("s_wait_asynccnt 0" ::: "memory");
#endif
}

// ---------------- persistent LSTM kernel ----------------
// 32 WGs x 128 threads. WG wg owns hidden units [wg*16, wg*16+16).
// Wave w (0..3) = gate w, computes gate columns n0..n0+15 for all 64 batch rows.
extern "C" __global__ __launch_bounds__(THR_)
void lstm_persistent(const _Float16* __restrict__ Wf,
                     const _Float16* __restrict__ xt,
                     const float*    __restrict__ bias,
                     _Float16*       __restrict__ h16,   // [2][B][H]
                     unsigned int*   __restrict__ cnt,   // [T]
                     float*          __restrict__ out) { // [B][T][H]
  extern __shared__ char smem[];
  _Float16* ldsW = (_Float16*)smem;                       // [64][832] f16  (106496 B)
  float*    ldsG = (float*)(smem + (size_t)64 * KT_ * 2); // [4][64][16] f32 (16384 B)
  float*    ldsC = ldsG + 4 * 64 * 16;                    // [64][16] f32    (4096 B)

  const int tid    = threadIdx.x;
  const int wg     = blockIdx.x;
  const int jbase  = wg * 16;
  const int wave   = tid >> 5;          // = gate index 0..3 (i,f,g,o)
  const int lane   = tid & 31;
  const int l15    = lane & 15;
  const int hiHalf = lane >> 4;
  const int gate   = wave;
  const int n0     = gate * H_ + jbase; // this wave's first gate column

  // ---- stage this WG's 64 rows of Wf into LDS via async global->LDS DMA ----
  // Dynamic LDS starts at offset 0 (no static __shared__ in this kernel), so the
  // LDS byte offset computed below is the VDST address for the async load.
  for (int c = tid; c < 64 * (KT_ / 8); c += THR_) {      // 16B chunks, 104 per row
    int r  = c / 104;
    int ci = c - r * 104;
    int n  = ((r >> 4) << 9) + jbase + (r & 15);          // (r/16)*512 + jbase + r%16
    unsigned ldsoff = (unsigned)(r * (KT_ * 2) + ci * 16);
    unsigned long long ga =
        (unsigned long long)(uintptr_t)(Wf + (size_t)n * KT_ + ci * 8);
    asm volatile("global_load_async_to_lds_b128 %0, %1, off"
                 :: "v"(ldsoff), "v"(ga) : "memory");
  }
  wait_asynccnt0();
  // ---- zero cell state slice ----
  for (int c = tid; c < 64 * 16; c += THR_) ldsC[c] = 0.f;
  const float biasv = bias[n0 + l15];
  __syncthreads();

  const _Float16* ldsRowB = ldsW + (size_t)(gate * 16 + l15) * KT_;

  for (int t = 0; t < T_; ++t) {
    // accumulators: 4 M-tiles (rows 0..63) x this wave's 16 columns, init = bias
    v8f acc0 = {biasv, biasv, biasv, biasv, biasv, biasv, biasv, biasv};
    v8f acc1 = acc0, acc2 = acc0, acc3 = acc0;

    // ---- x-part GEMM (independent of h(t-1): overlaps the barrier wait) ----
    const _Float16* xtt = xt + (size_t)t * B_ * IP_;
    for (int ks = 0; ks < IP_ / 32; ++ks) {
      int k0 = ks * 32;
      v16h bm = loadB32x16_lds(ldsRowB, H_ + k0, hiHalf);
      v16h a0 = loadA16x32(xtt, IP_,  0 + l15, k0, hiHalf);
      acc0 = WMMA_F16(a0, bm, acc0);
      v16h a1 = loadA16x32(xtt, IP_, 16 + l15, k0, hiHalf);
      acc1 = WMMA_F16(a1, bm, acc1);
      v16h a2 = loadA16x32(xtt, IP_, 32 + l15, k0, hiHalf);
      acc2 = WMMA_F16(a2, bm, acc2);
      v16h a3 = loadA16x32(xtt, IP_, 48 + l15, k0, hiHalf);
      acc3 = WMMA_F16(a3, bm, acc3);
    }

    // ---- prefetch next timestep's x tile; latency hides under the spin-wait ----
    if (t + 1 < T_) {
      const char* nx = (const char*)(xt + (size_t)(t + 1) * B_ * IP_);
      for (int o = tid; o < (B_ * IP_ * 2) / 128; o += THR_)
        __builtin_prefetch(nx + (size_t)o * 128, 0, 0);
    }

    // ---- wait until every WG published h(t-1) ----
    if (t > 0) {
      while (__hip_atomic_load(&cnt[t - 1], __ATOMIC_ACQUIRE,
                               __HIP_MEMORY_SCOPE_AGENT) < NWG_)
        __builtin_amdgcn_s_sleep(1);
    }

    // ---- h-part GEMM: read h(t-1) from buffer (1 - t&1) ----
    const _Float16* hr = h16 + (size_t)(1 - (t & 1)) * B_ * H_;
    for (int ks = 0; ks < H_ / 32; ++ks) {
      int k0 = ks * 32;
      v16h bm = loadB32x16_lds(ldsRowB, k0, hiHalf);
      v16h a0 = loadA16x32(hr, H_,  0 + l15, k0, hiHalf);
      acc0 = WMMA_F16(a0, bm, acc0);
      v16h a1 = loadA16x32(hr, H_, 16 + l15, k0, hiHalf);
      acc1 = WMMA_F16(a1, bm, acc1);
      v16h a2 = loadA16x32(hr, H_, 32 + l15, k0, hiHalf);
      acc2 = WMMA_F16(a2, bm, acc2);
      v16h a3 = loadA16x32(hr, H_, 48 + l15, k0, hiHalf);
      acc3 = WMMA_F16(a3, bm, acc3);
    }

    // ---- scatter gate tiles to LDS (C layout: vgpr v -> row v + 8*hiHalf) ----
    {
      float* g = ldsG + wave * 1024;  // [64][16]
      v8f a[4] = {acc0, acc1, acc2, acc3};
#pragma unroll
      for (int mt = 0; mt < 4; ++mt)
#pragma unroll
        for (int v = 0; v < 8; ++v)
          g[(mt * 16 + v + (hiHalf << 3)) * 16 + l15] = a[mt][v];
    }
    __syncthreads();

    // ---- elementwise LSTM cell update for this WG's 64x16 slice ----
    _Float16* hw = h16 + (size_t)(t & 1) * B_ * H_;
#pragma unroll
    for (int u = 0; u < 8; ++u) {
      int idx = tid + u * THR_;        // 0..1023
      int b  = idx >> 4;
      int jj = idx & 15;
      float gi = ldsG[0 * 1024 + b * 16 + jj];
      float gf = ldsG[1 * 1024 + b * 16 + jj];
      float gg = ldsG[2 * 1024 + b * 16 + jj];
      float go = ldsG[3 * 1024 + b * 16 + jj];
      float iv = sigmoidf_(gi);
      float fv = sigmoidf_(gf);
      float gv = tanhf_(gg);
      float ov = sigmoidf_(go);
      float cv = fv * ldsC[idx] + iv * gv;
      ldsC[idx] = cv;
      float hv = ov * tanhf_(cv);
      out[((size_t)b * T_ + t) * H_ + jbase + jj] = hv;
      hw[b * H_ + jbase + jj] = (_Float16)hv;
    }
    __syncthreads();

    // ---- publish h(t) ----
    if (tid == 0)
      __hip_atomic_fetch_add(&cnt[t], 1u, __ATOMIC_RELEASE,
                             __HIP_MEMORY_SCOPE_AGENT);
  }
}

// ---------------- host launch ----------------
extern "C" void kernel_launch(void* const* d_in, const int* in_sizes, int n_in,
                              void* d_out, int out_size, void* d_ws, size_t ws_size,
                              hipStream_t stream) {
  const float* x   = (const float*)d_in[0];
  const float* Wih = (const float*)d_in[1];
  const float* Whh = (const float*)d_in[2];
  const float* bih = (const float*)d_in[3];
  const float* bhh = (const float*)d_in[4];

  char* ws = (char*)d_ws;
  _Float16* Wf   = (_Float16*)(ws + WF_OFF);
  _Float16* xt   = (_Float16*)(ws + XT_OFF);
  float*    bias = (float*)   (ws + BIAS_OFF);
  _Float16* h16  = (_Float16*)(ws + H16_OFF);
  unsigned* cnt  = (unsigned*)(ws + CNT_OFF);

  // zero h double-buffer + per-step barrier counters (graph-capturable)
  (void)hipMemsetAsync(ws + H16_OFF, 0, (size_t)(H16_BYTES + CNT_BYTES), stream);

  lstm_prep_weights<<<(G4_ * KT_ + 255) / 256, 256, 0, stream>>>(Whh, Wih, bih, bhh, Wf, bias);
  lstm_prep_x<<<(int)(((long long)T_ * B_ * IP_ + 255) / 256), 256, 0, stream>>>(x, xt);

  size_t lds = (size_t)64 * KT_ * 2 + (size_t)4 * 64 * 16 * 4 + (size_t)64 * 16 * 4;
  lstm_persistent<<<NWG_, THR_, lds, stream>>>(Wf, xt, bias, h16, cnt, (float*)d_out);
}
